// SwinTransformerAttention_32160715113211
// MI455X (gfx1250) — matrix operand order repaired and checked by reading, compiled, not run on hardware
//
#include <hip/hip_runtime.h>
#include <hip/hip_bf16.h>

// ---------------------------------------------------------------------------
// Swin window attention for MI455X (gfx1250, wave32, WMMA 16x16x32 bf16).
//
//   attn = softmax( w^T (Wq^T Wk) w  + colterm(m) + relposbias(n,m) )
//   out  = Wp * windowreverse( attn * (Wv w + bv) ) + bp
//
// (row term Wq^T bk . w_n and constant bq.bk are softmax-invariant -> dropped)
// ---------------------------------------------------------------------------

#define CDIM   192
#define ODIM   192
#define HDIM   224
#define NPIX   49
#define PSTRIDE 50176          // 224*224
#define NWINB  8192            // 8 * 32 * 32 windows

typedef __attribute__((ext_vector_type(16))) __bf16 bf16x16;
typedef __attribute__((ext_vector_type(8)))  __bf16 bf16x8;
typedef __attribute__((ext_vector_type(8)))  float  floatx8;

// workspace layout (byte offsets, all 256-aligned)
#define OFF_GT   0u            // 192*192 bf16 : Gt[c2][c] = sum_o Wq[o][c]*Wk[o][c2]
#define OFF_WV   73728u        // 192*192 bf16 : Wv row-major
#define OFF_WP   147456u       // 192*192 bf16 : Wp row-major
#define OFF_B2   221184u       // 192 f32     : b2[c] = sum_o bq[o]*Wk[o][c]
#define OFF_BIAS 222208u       // 64*49 f32   : rel-pos bias, rows 49..63 zero
#define OFF_Y    234752u       // 8*192*50176 f32 : pre-projection activations

__device__ __forceinline__ floatx8 wmma_bf16(bf16x16 a, bf16x16 b, floatx8 c) {
  // D = A(16x32 bf16) * B(32x16 bf16) + C(16x16 f32)
  return __builtin_amdgcn_wmma_f32_16x16x32_bf16(false, a, false, b,
                                                 (short)0, c, false, false);
}

// A fragment: lane m = lane%16 fixed; k = hf*8+0..7 (VGPR0-3), 16+hf*8+0..7 (VGPR4-7)
__device__ __forceinline__ bf16x16 fragA(const __bf16* p, int hf) {
  bf16x8 lo = *(const bf16x8*)(p + hf * 8);
  bf16x8 hi = *(const bf16x8*)(p + 16 + hf * 8);
  return __builtin_shufflevector(lo, hi, 0,1,2,3,4,5,6,7,8,9,10,11,12,13,14,15);
}
// B fragment: lane n = lane%16 fixed; k = hf*8+0..7 (VGPR0-3), 16+hf*8+0..7 (VGPR4-7)
__device__ __forceinline__ bf16x16 fragB(const __bf16* p, int hf) {
  bf16x8 lo = *(const bf16x8*)(p + hf * 8);
  bf16x8 hi = *(const bf16x8*)(p + 16 + hf * 8);
  return __builtin_shufflevector(lo, hi, 0,1,2,3,4,5,6,7,8,9,10,11,12,13,14,15);
}

// ---------------------------------------------------------------------------
// Prep 1: Gt (bf16, transposed for B-fragment loads) and b2 vector.
// ---------------------------------------------------------------------------
__global__ void swin_prep_gt(const float* __restrict__ Wq,
                             const float* __restrict__ Wk,
                             const float* __restrict__ bq,
                             __bf16* __restrict__ Gt, float* __restrict__ b2) {
  int blk = blockIdx.x;
  int t = threadIdx.x;
  if (t >= CDIM) return;
  if (blk < CDIM) {
    float s = 0.f;
    for (int o = 0; o < ODIM; ++o) s += Wq[o * CDIM + t] * Wk[o * CDIM + blk];
    Gt[blk * CDIM + t] = (__bf16)s;
  } else {
    float s = 0.f;
    for (int o = 0; o < ODIM; ++o) s += bq[o] * Wk[o * CDIM + t];
    b2[t] = s;
  }
}

// ---------------------------------------------------------------------------
// Prep 2: bf16 copies of Wv, Wp; 49x49 relative-position bias (padded to 64).
// ---------------------------------------------------------------------------
__global__ void swin_prep_misc(const float* __restrict__ Wv,
                               const float* __restrict__ Wp,
                               const float* __restrict__ btab,
                               __bf16* __restrict__ WvB, __bf16* __restrict__ WpB,
                               float* __restrict__ bias49) {
  int idx = blockIdx.x * 256 + threadIdx.x;
  if (idx < CDIM * ODIM) { WvB[idx] = (__bf16)Wv[idx]; return; }
  idx -= CDIM * ODIM;
  if (idx < CDIM * ODIM) { WpB[idx] = (__bf16)Wp[idx]; return; }
  idx -= CDIM * ODIM;
  if (idx < 64 * NPIX) {
    int n = idx / NPIX, m = idx - (idx / NPIX) * NPIX;
    float r = 0.f;
    if (n < NPIX) {
      int yi = n / 7, xi = n - yi * 7, yj = m / 7, xj = m - yj * 7;
      int ri = (yi - yj + 6) * 13 + (xi - xj + 6);
      float s = 0.f;
      for (int o = 0; o < ODIM; ++o) s += btab[ri * ODIM + o];
      r = s * (1.f / 192.f);
    }
    bias49[idx] = r;
  }
}

// ---------------------------------------------------------------------------
// Fused per-window attention. One block = one 7x7 window, 8 waves.
// LDS: wT[64][200] | (r[64][216] then vT[192][72]) | attn_bf16[64][72] | colterm[64]
// ---------------------------------------------------------------------------
__global__ __launch_bounds__(256) void swin_attn(
    const float* __restrict__ x, const __bf16* __restrict__ Gt,
    const __bf16* __restrict__ WvB, const float* __restrict__ b2,
    const float* __restrict__ bias49, const float* __restrict__ bv,
    float* __restrict__ y) {
  __shared__ __align__(16) char smem[62720];
  __bf16* sWT = (__bf16*)smem;                 // [64][200] bf16  (25600 B)
  __bf16* sRV = (__bf16*)(smem + 25600);       // r [64][216] -> vT [192][72] (27648 B)
  __bf16* sAt = (__bf16*)(smem + 53248);       // attn bf16 [64][72] (9216 B)
  float*  sCT = (float*)(smem + 62464);        // colterm[64]

  const int tid  = threadIdx.x;
  const int wave = tid >> 5;
  const int lane = tid & 31;
  const int ln   = lane & 15;
  const int hf   = lane >> 4;

  const int bw = blockIdx.x;
  const int b  = bw >> 10;
  const int wh = (bw >> 5) & 31;
  const int ww = bw & 31;

  // ---- Phase 1: gather cyclically shifted window, transposed, f32 -> bf16 --
  const float* xb = x + (size_t)b * CDIM * PSTRIDE;
  for (int idx = tid; idx < CDIM * NPIX; idx += 256) {
    int c = idx / NPIX;
    int n = idx - c * NPIX;
    int wy = n / 7, wx = n - wy * 7;
    int h = wh * 7 + wy + 3; if (h >= HDIM) h -= HDIM;   // shift -3 then window
    int w = ww * 7 + wx + 3; if (w >= HDIM) w -= HDIM;
    sWT[n * 200 + c] = (__bf16)xb[(size_t)c * PSTRIDE + h * HDIM + w];
  }
  for (int idx = tid; idx < 15 * 200; idx += 256)        // zero pad rows 49..63
    sWT[NPIX * 200 + idx] = (__bf16)0.f;
  __syncthreads();

  // ---- Phase 2: r = wT * G   (48 tiles, 6 per wave), plus colterm ---------
  for (int t6 = 0; t6 < 6; ++t6) {
    int tile = wave + 8 * t6;                  // 0..47
    int mt = tile / 12, ot = tile - mt * 12;
    floatx8 acc = {0.f,0.f,0.f,0.f,0.f,0.f,0.f,0.f};
    const __bf16* pa = sWT + (mt * 16 + ln) * 200;
    const __bf16* pb = Gt + (ot * 16 + ln) * CDIM;
#pragma unroll
    for (int kc = 0; kc < 6; ++kc)
      acc = wmma_bf16(fragA(pa + kc * 32, hf), fragB(pb + kc * 32, hf), acc);
#pragma unroll
    for (int i = 0; i < 8; ++i)
      sRV[(mt * 16 + i + 8 * hf) * 216 + ot * 16 + ln] = (__bf16)acc[i];
  }
  if (tid < 64) {                              // colterm[m] = b2 . w[:,m]
    float s = 0.f;
    if (tid < NPIX)
      for (int c = 0; c < CDIM; ++c) s += b2[c] * (float)sWT[tid * 200 + c];
    sCT[tid] = s;
  }
  __syncthreads();

  // ---- Phase 3: attn = r * w   (waves 0..3, one 16-row band each) ---------
  floatx8 at[4];
  if (wave < 4) {
    const __bf16* pa = sRV + (wave * 16 + ln) * 216;
#pragma unroll
    for (int t = 0; t < 4; ++t) {
      floatx8 acc = {0.f,0.f,0.f,0.f,0.f,0.f,0.f,0.f};
      const __bf16* pb = sWT + (t * 16 + ln) * 200;   // B = w[c'][m] via wT rows
#pragma unroll
      for (int kc = 0; kc < 6; ++kc)
        acc = wmma_bf16(fragA(pa + kc * 32, hf), fragB(pb + kc * 32, hf), acc);
      at[t] = acc;
    }
  }
  __syncthreads();   // sRV (r) now dead -> reusable as vT

  if (wave < 4) {
    // register softmax: row = wave*16 + i + 8*hf, its 16 cols live across lanes
    float rm[8];
#pragma unroll
    for (int i = 0; i < 8; ++i) rm[i] = -3.0e38f;
#pragma unroll
    for (int t = 0; t < 4; ++t) {
      int col = t * 16 + ln;
#pragma unroll
      for (int i = 0; i < 8; ++i) {
        int row = wave * 16 + i + 8 * hf;
        float v = at[t][i];
        if (col < NPIX) v += bias49[row * NPIX + col] + sCT[col];
        else            v = -3.0e38f;          // mask padded columns
        at[t][i] = v;
        rm[i] = fmaxf(rm[i], v);
      }
    }
#pragma unroll
    for (int i = 0; i < 8; ++i) {
      float m = rm[i];
      m = fmaxf(m, __shfl_xor(m, 1));
      m = fmaxf(m, __shfl_xor(m, 2));
      m = fmaxf(m, __shfl_xor(m, 4));
      m = fmaxf(m, __shfl_xor(m, 8));
      rm[i] = m;
    }
    float sm[8];
#pragma unroll
    for (int i = 0; i < 8; ++i) sm[i] = 0.f;
#pragma unroll
    for (int t = 0; t < 4; ++t)
#pragma unroll
      for (int i = 0; i < 8; ++i) {
        float e = __expf(at[t][i] - rm[i]);
        at[t][i] = e;
        sm[i] += e;
      }
#pragma unroll
    for (int i = 0; i < 8; ++i) {
      float s = sm[i];
      s += __shfl_xor(s, 1);
      s += __shfl_xor(s, 2);
      s += __shfl_xor(s, 4);
      s += __shfl_xor(s, 8);
      sm[i] = (s > 0.f) ? (1.f / s) : 0.f;
    }
#pragma unroll
    for (int t = 0; t < 4; ++t)
#pragma unroll
      for (int i = 0; i < 8; ++i)
        sAt[(wave * 16 + i + 8 * hf) * 72 + t * 16 + ln] = (__bf16)(at[t][i] * sm[i]);
  }

  // ---- Phase 4: v = wT * Wv + bv, stored transposed into sRV --------------
  for (int t6 = 0; t6 < 6; ++t6) {
    int tile = wave + 8 * t6;
    int mt = tile / 12, ot = tile - mt * 12;
    floatx8 acc = {0.f,0.f,0.f,0.f,0.f,0.f,0.f,0.f};
    const __bf16* pa = sWT + (mt * 16 + ln) * 200;
    const __bf16* pb = WvB + (ot * 16 + ln) * CDIM;
#pragma unroll
    for (int kc = 0; kc < 6; ++kc)
      acc = wmma_bf16(fragA(pa + kc * 32, hf), fragB(pb + kc * 32, hf), acc);
    float bvv = bv[ot * 16 + ln];              // col o fixed per lane
#pragma unroll
    for (int i = 0; i < 8; ++i)
      sRV[(ot * 16 + ln) * 72 + mt * 16 + i + 8 * hf] = (__bf16)(acc[i] + bvv);
  }
  __syncthreads();

  // ---- Phase 5: out = attn * v, scatter with reverse shift ----------------
  for (int t6 = 0; t6 < 6; ++t6) {
    int tile = wave + 8 * t6;
    int mt = tile / 12, ot = tile - mt * 12;
    floatx8 acc = {0.f,0.f,0.f,0.f,0.f,0.f,0.f,0.f};
    const __bf16* pa = sAt + (mt * 16 + ln) * 72;
    const __bf16* pb = sRV + (ot * 16 + ln) * 72;   // vT rows
#pragma unroll
    for (int kc = 0; kc < 2; ++kc)
      acc = wmma_bf16(fragA(pa + kc * 32, hf), fragB(pb + kc * 32, hf), acc);
    int o = ot * 16 + ln;
#pragma unroll
    for (int i = 0; i < 8; ++i) {
      int n = mt * 16 + i + 8 * hf;
      if (n < NPIX) {
        int wy = n / 7, wx = n - wy * 7;
        int h = wh * 7 + wy + 3; if (h >= HDIM) h -= HDIM;  // reverse shift +3
        int w = ww * 7 + wx + 3; if (w >= HDIM) w -= HDIM;
        y[((size_t)(b * ODIM + o)) * PSTRIDE + h * HDIM + w] = acc[i];
      }
    }
  }
}

// ---------------------------------------------------------------------------
// Final 1x1 conv: out = Wp * y + bp. One block = one image x 128 pixels.
// ---------------------------------------------------------------------------
__global__ __launch_bounds__(256) void swin_proj(
    const float* __restrict__ y, const __bf16* __restrict__ WpB,
    const float* __restrict__ bp, float* __restrict__ out) {
  __shared__ __align__(16) __bf16 sYT[128 * 200];   // yT[p][c], 51200 B

  const int tid  = threadIdx.x;
  const int wave = tid >> 5;
  const int lane = tid & 31;
  const int ln   = lane & 15;
  const int hf   = lane >> 4;

  const int blk = blockIdx.x;
  const int b   = blk / 392;
  const int p0  = (blk - b * 392) * 128;

  const float* yb = y + (size_t)b * CDIM * PSTRIDE;
  for (int idx = tid; idx < CDIM * 32; idx += 256) {
    int c = idx >> 5;
    int g = idx & 31;
    float4 v4 = *(const float4*)(yb + (size_t)c * PSTRIDE + p0 + g * 4);
    int pl = g * 4;
    sYT[(pl + 0) * 200 + c] = (__bf16)v4.x;
    sYT[(pl + 1) * 200 + c] = (__bf16)v4.y;
    sYT[(pl + 2) * 200 + c] = (__bf16)v4.z;
    sYT[(pl + 3) * 200 + c] = (__bf16)v4.w;
  }
  __syncthreads();

  for (int t12 = 0; t12 < 12; ++t12) {
    int tile = wave + 8 * t12;                 // 0..95
    int ot = tile >> 3, pt = tile & 7;
    floatx8 acc = {0.f,0.f,0.f,0.f,0.f,0.f,0.f,0.f};
    const __bf16* pa = WpB + (ot * 16 + ln) * CDIM;
    const __bf16* pb = sYT + (pt * 16 + ln) * 200;
#pragma unroll
    for (int kc = 0; kc < 6; ++kc)
      acc = wmma_bf16(fragA(pa + kc * 32, hf), fragB(pb + kc * 32, hf), acc);
#pragma unroll
    for (int i = 0; i < 8; ++i) {
      int o = ot * 16 + i + 8 * hf;
      out[((size_t)(b * ODIM + o)) * PSTRIDE + p0 + pt * 16 + ln] = acc[i] + bp[o];
    }
  }
}

// ---------------------------------------------------------------------------
extern "C" void kernel_launch(void* const* d_in, const int* in_sizes, int n_in,
                              void* d_out, int out_size, void* d_ws, size_t ws_size,
                              hipStream_t stream) {
  const float* x    = (const float*)d_in[0];
  const float* Wq   = (const float*)d_in[1];
  const float* bq   = (const float*)d_in[2];
  const float* Wk   = (const float*)d_in[3];
  // d_in[4] = bk: contributes only a per-row constant -> softmax-invariant, unused
  const float* Wv   = (const float*)d_in[5];
  const float* bv   = (const float*)d_in[6];
  const float* Wp   = (const float*)d_in[7];
  const float* bp   = (const float*)d_in[8];
  const float* btab = (const float*)d_in[9];
  (void)in_sizes; (void)n_in; (void)out_size; (void)ws_size;

  char* ws = (char*)d_ws;
  __bf16* Gt     = (__bf16*)(ws + OFF_GT);
  __bf16* WvB    = (__bf16*)(ws + OFF_WV);
  __bf16* WpB    = (__bf16*)(ws + OFF_WP);
  float*  b2     = (float*)(ws + OFF_B2);
  float*  bias49 = (float*)(ws + OFF_BIAS);
  float*  y      = (float*)(ws + OFF_Y);
  float*  outp   = (float*)d_out;

  swin_prep_gt<<<193, 256, 0, stream>>>(Wq, Wk, bq, Gt, b2);
  swin_prep_misc<<<301, 256, 0, stream>>>(Wv, Wp, btab, WvB, WpB, bias49);
  swin_attn<<<NWINB, 256, 0, stream>>>(x, Gt, WvB, b2, bias49, bv, y);
  swin_proj<<<3136, 256, 0, stream>>>(y, WpB, bp, outp);
}
